// DecoderBlock_47390669144299
// MI455X (gfx1250) — matrix operand order repaired and checked by reading, not compiled, not run on hardware
//
#include <hip/hip_runtime.h>
#include <hip/hip_bf16.h>

typedef __bf16 bf16;
typedef __attribute__((ext_vector_type(16))) __bf16 bf16x16;
typedef __attribute__((ext_vector_type(8)))  __bf16 bf16x8;
typedef __attribute__((ext_vector_type(8)))  float  v8f;

#define TSEQ   4096
#define CEMB   768
#define DHEAD  64
#define NHEADS 12
#define FFND   3072

#define WMMA_BF16(a, b, c) \
  __builtin_amdgcn_wmma_f32_16x16x32_bf16(false, (a), false, (b), (short)0, (c), false, false)

// ---------------- fragment loaders ----------------
// A-matrix 16x32 bf16 layout (ISA 7.12.2): lane<16 -> row=lane, K=0..7 & 16..23
//                                          lane>=16 -> row=lane-16, K=8..15 & 24..31
__device__ __forceinline__ bf16x16 load_a_frag(const bf16* base, int stride, int lane) {
  const bf16* p = base + (size_t)(lane & 15) * stride + ((lane >> 4) << 3);
  bf16x8 lo = *(const bf16x8*)p;
  bf16x8 hi = *(const bf16x8*)(p + 16);
  bf16x16 r;
#pragma unroll
  for (int i = 0; i < 8; ++i) { r[i] = lo[i]; r[i + 8] = hi[i]; }
  return r;
}

// B-matrix 32x16 bf16 layout: lane<16 -> col=lane, K=0..7 & 16..23 ; lane>=16 -> col=lane-16, K=8..15 & 24..31
// base points at BT[col0][k0] with BT stored [N][K] (contiguous K)
__device__ __forceinline__ bf16x16 load_b_frag(const bf16* base, int stride, int lane) {
  const bf16* p = base + (size_t)(lane & 15) * stride + ((lane >> 4) << 3);
  bf16x8 lo = *(const bf16x8*)p;
  bf16x8 hi = *(const bf16x8*)(p + 16);
  bf16x16 r;
#pragma unroll
  for (int i = 0; i < 8; ++i) { r[i] = lo[i]; r[i + 8] = hi[i]; }
  return r;
}

// ---------------- double-buffered 32x64 per-wave GEMM tile ----------------
// acc[0..3] = rows m0..m0+15 x cols n0+{0,16,32,48}; acc[4..7] = rows m0+16..m0+31
template <int KD>
__device__ __forceinline__ void gemm_tile_32x64(const bf16* __restrict__ A,
                                                const bf16* __restrict__ BT,
                                                int m0, int n0, int lane, v8f acc[8]) {
  const bf16* a0p = A + (size_t)m0 * KD;
  const bf16* a1p = A + (size_t)(m0 + 16) * KD;
  const bf16* bp  = BT + (size_t)n0 * KD;
  bf16x16 A0[2], A1[2], B0[2], B1[2], B2[2], B3[2];
  A0[0] = load_a_frag(a0p, KD, lane);
  A1[0] = load_a_frag(a1p, KD, lane);
  B0[0] = load_b_frag(bp + (size_t)0 * KD, KD, lane);
  B1[0] = load_b_frag(bp + (size_t)16 * KD, KD, lane);
  B2[0] = load_b_frag(bp + (size_t)32 * KD, KD, lane);
  B3[0] = load_b_frag(bp + (size_t)48 * KD, KD, lane);
#pragma unroll 1
  for (int k0 = 0; k0 < KD; k0 += 64) {
    // preload odd set (k0+32) -- always in range since KD % 64 == 0
    A0[1] = load_a_frag(a0p + k0 + 32, KD, lane);
    A1[1] = load_a_frag(a1p + k0 + 32, KD, lane);
    B0[1] = load_b_frag(bp + (size_t)0 * KD + k0 + 32, KD, lane);
    B1[1] = load_b_frag(bp + (size_t)16 * KD + k0 + 32, KD, lane);
    B2[1] = load_b_frag(bp + (size_t)32 * KD + k0 + 32, KD, lane);
    B3[1] = load_b_frag(bp + (size_t)48 * KD + k0 + 32, KD, lane);
    // compute on even set (loads above are independent -> overlap)
    acc[0] = WMMA_BF16(A0[0], B0[0], acc[0]);
    acc[1] = WMMA_BF16(A0[0], B1[0], acc[1]);
    acc[2] = WMMA_BF16(A0[0], B2[0], acc[2]);
    acc[3] = WMMA_BF16(A0[0], B3[0], acc[3]);
    acc[4] = WMMA_BF16(A1[0], B0[0], acc[4]);
    acc[5] = WMMA_BF16(A1[0], B1[0], acc[5]);
    acc[6] = WMMA_BF16(A1[0], B2[0], acc[6]);
    acc[7] = WMMA_BF16(A1[0], B3[0], acc[7]);
    // preload even set for next iteration (guarded: no OOB on final step)
    if (k0 + 64 < KD) {
      A0[0] = load_a_frag(a0p + k0 + 64, KD, lane);
      A1[0] = load_a_frag(a1p + k0 + 64, KD, lane);
      B0[0] = load_b_frag(bp + (size_t)0 * KD + k0 + 64, KD, lane);
      B1[0] = load_b_frag(bp + (size_t)16 * KD + k0 + 64, KD, lane);
      B2[0] = load_b_frag(bp + (size_t)32 * KD + k0 + 64, KD, lane);
      B3[0] = load_b_frag(bp + (size_t)48 * KD + k0 + 64, KD, lane);
    }
    // compute on odd set
    acc[0] = WMMA_BF16(A0[1], B0[1], acc[0]);
    acc[1] = WMMA_BF16(A0[1], B1[1], acc[1]);
    acc[2] = WMMA_BF16(A0[1], B2[1], acc[2]);
    acc[3] = WMMA_BF16(A0[1], B3[1], acc[3]);
    acc[4] = WMMA_BF16(A1[1], B0[1], acc[4]);
    acc[5] = WMMA_BF16(A1[1], B1[1], acc[5]);
    acc[6] = WMMA_BF16(A1[1], B2[1], acc[6]);
    acc[7] = WMMA_BF16(A1[1], B3[1], acc[7]);
  }
}

__device__ __forceinline__ void zero_acc8(v8f acc[8]) {
#pragma unroll
  for (int t = 0; t < 8; ++t) {
#pragma unroll
    for (int r = 0; r < 8; ++r) acc[t][r] = 0.0f;
  }
}

// ---------------- weight convert: fp32 [din][dout] -> bf16 [dout][din] ----------------
__global__ __launch_bounds__(256) void convert_wT(const float* __restrict__ W,
                                                  bf16* __restrict__ WT,
                                                  int din, int dout) {
  int idx = blockIdx.x * 256 + threadIdx.x;
  if (idx >= din * dout) return;
  int n = idx / din;
  int k = idx - n * din;
  WT[idx] = (bf16)W[(size_t)k * dout + n];
}

// ---------------- LayerNorm: one wave per row of 768 ----------------
__global__ __launch_bounds__(256) void ln_kernel(const float* __restrict__ x,
                                                 const float* __restrict__ g,
                                                 const float* __restrict__ b,
                                                 float* __restrict__ of,
                                                 bf16* __restrict__ ob) {
  int lane = threadIdx.x & 31;
  int row = blockIdx.x * 8 + (threadIdx.x >> 5);
  const float* xr = x + (size_t)row * CEMB;
  float v[24];
  float s = 0.0f;
#pragma unroll
  for (int i = 0; i < 24; ++i) { v[i] = xr[lane + 32 * i]; s += v[i]; }
#pragma unroll
  for (int m = 16; m >= 1; m >>= 1) s += __shfl_xor(s, m, 32);
  float mu = s * (1.0f / 768.0f);
  float s2 = 0.0f;
#pragma unroll
  for (int i = 0; i < 24; ++i) { float d = v[i] - mu; s2 += d * d; }
#pragma unroll
  for (int m = 16; m >= 1; m >>= 1) s2 += __shfl_xor(s2, m, 32);
  float rstd = rsqrtf(s2 * (1.0f / 768.0f) + 1e-5f);
#pragma unroll
  for (int i = 0; i < 24; ++i) {
    int c = lane + 32 * i;
    float y = (v[i] - mu) * rstd * g[c] + b[c];
    of[(size_t)row * CEMB + c] = y;
    ob[(size_t)row * CEMB + c] = (bf16)y;
  }
}

// ---------------- QKV: h @ [Wq|Wk|Wv] + bias, scatter to Q[h][t][d], K[h][t][d], VT[h][d][t] ----------------
__global__ __launch_bounds__(256) void qkv_kernel(const bf16* __restrict__ H,
                                                  const bf16* __restrict__ WT,
                                                  const float* __restrict__ bq,
                                                  const float* __restrict__ bk,
                                                  const float* __restrict__ bv,
                                                  bf16* __restrict__ Q,
                                                  bf16* __restrict__ K,
                                                  bf16* __restrict__ VT) {
  int lane = threadIdx.x & 31;
  int w = blockIdx.x * 8 + (threadIdx.x >> 5);
  const int ntn = (3 * CEMB) / 64;  // 36
  int m0 = (w / ntn) * 32;
  int n0 = (w % ntn) * 64;
  v8f acc[8];
  zero_acc8(acc);
  gemm_tile_32x64<CEMB>(H, WT, m0, n0, lane, acc);
  // fold (1/sqrt(D)) * log2(e) into Q so attention can use exp2
  const float QS = 0.125f * 1.44269504088896340736f;
  int hi = lane >> 4, lo = lane & 15;
#pragma unroll
  for (int ms = 0; ms < 2; ++ms) {
#pragma unroll
    for (int t = 0; t < 4; ++t) {
#pragma unroll
      for (int r = 0; r < 8; ++r) {
        int row = m0 + ms * 16 + r + 8 * hi;
        int n = n0 + t * 16 + lo;
        int reg = n / CEMB;
        int nn = n - reg * CEMB;
        int hh = nn >> 6, dd = nn & 63;
        float val = acc[ms * 4 + t][r];
        if (reg == 0) {
          val += bq[nn];
          Q[((size_t)hh * TSEQ + row) * DHEAD + dd] = (bf16)(val * QS);
        } else if (reg == 1) {
          val += bk[nn];
          K[((size_t)hh * TSEQ + row) * DHEAD + dd] = (bf16)val;
        } else {
          val += bv[nn];
          VT[((size_t)hh * DHEAD + dd) * TSEQ + row] = (bf16)val;
        }
      }
    }
  }
}

// ---------------- Flash attention: one wave per (head, 16-query tile) ----------------
__global__ __launch_bounds__(256) void attn_kernel(const bf16* __restrict__ Q,
                                                   const bf16* __restrict__ K,
                                                   const bf16* __restrict__ VT,
                                                   bf16* __restrict__ ctx) {
  __shared__ bf16 plds[8 * 512];  // per-wave 16x32 bf16 P tile
  int lane = threadIdx.x & 31;
  int wv = threadIdx.x >> 5;
  int wid = blockIdx.x * 8 + wv;
  int h = wid >> 8;  // / 256 query tiles
  int qt = wid & 255;
  int q0 = qt * 16;
  int hi = lane >> 4, lo = lane & 15;
  const bf16* Qh = Q + (size_t)h * TSEQ * DHEAD;
  const bf16* Kh = K + (size_t)h * TSEQ * DHEAD;
  const bf16* Vh = VT + (size_t)h * DHEAD * TSEQ;
  bf16* myp = plds + wv * 512;

  bf16x16 qf0 = load_a_frag(Qh + (size_t)q0 * DHEAD + 0, DHEAD, lane);
  bf16x16 qf1 = load_a_frag(Qh + (size_t)q0 * DHEAD + 32, DHEAD, lane);

  float mrow[8], lrow[8];
  v8f acc[4];
#pragma unroll
  for (int t = 0; t < 4; ++t) {
#pragma unroll
    for (int r = 0; r < 8; ++r) acc[t][r] = 0.0f;
  }
#pragma unroll
  for (int r = 0; r < 8; ++r) { mrow[r] = -3.0e38f; lrow[r] = 0.0f; }

  int nk = q0 + 16;  // causal: keys 0..q0+15
#pragma unroll 1
  for (int kb = 0; kb < nk; kb += 32) {
    // issue ALL loads for this 32-key step up front: 4 K-fragments + 4 V-fragments.
    // V-fragments are consumed only after the softmax VALU section -> long overlap.
    bf16x16 kf00 = load_b_frag(Kh + (size_t)kb * DHEAD + 0, DHEAD, lane);
    bf16x16 kf01 = load_b_frag(Kh + (size_t)kb * DHEAD + 32, DHEAD, lane);
    bf16x16 kf10 = load_b_frag(Kh + (size_t)(kb + 16) * DHEAD + 0, DHEAD, lane);
    bf16x16 kf11 = load_b_frag(Kh + (size_t)(kb + 16) * DHEAD + 32, DHEAD, lane);
    bf16x16 vf[4];
#pragma unroll
    for (int t = 0; t < 4; ++t)
      vf[t] = load_b_frag(Vh + (size_t)(t * 16) * TSEQ + kb, TSEQ, lane);

    v8f sc[2];
#pragma unroll
    for (int j = 0; j < 2; ++j) {
      v8f s;
#pragma unroll
      for (int r = 0; r < 8; ++r) s[r] = 0.0f;
      s = WMMA_BF16(qf0, (j ? kf10 : kf00), s);
      s = WMMA_BF16(qf1, (j ? kf11 : kf01), s);
      int kbj = kb + j * 16;
#pragma unroll
      for (int r = 0; r < 8; ++r) {
        int row = q0 + r + 8 * hi;
        int key = kbj + lo;
        if (key > row) s[r] = -3.0e38f;  // causal mask
      }
      sc[j] = s;
    }
    // online softmax update (per-row stats replicated across each 16-lane group)
#pragma unroll
    for (int r = 0; r < 8; ++r) {
      float x = fmaxf(sc[0][r], sc[1][r]);
      x = fmaxf(x, __shfl_xor(x, 1, 32));
      x = fmaxf(x, __shfl_xor(x, 2, 32));
      x = fmaxf(x, __shfl_xor(x, 4, 32));
      x = fmaxf(x, __shfl_xor(x, 8, 32));
      float mnew = fmaxf(mrow[r], x);
      float scal = exp2f(mrow[r] - mnew);
      float p0 = exp2f(sc[0][r] - mnew);
      float p1 = exp2f(sc[1][r] - mnew);
      float rs = p0 + p1;
      rs += __shfl_xor(rs, 1, 32);
      rs += __shfl_xor(rs, 2, 32);
      rs += __shfl_xor(rs, 4, 32);
      rs += __shfl_xor(rs, 8, 32);
      lrow[r] = lrow[r] * scal + rs;
      mrow[r] = mnew;
#pragma unroll
      for (int t = 0; t < 4; ++t) acc[t][r] *= scal;
      int prow = r + 8 * hi;
      myp[prow * 32 + lo] = (bf16)p0;
      myp[prow * 32 + 16 + lo] = (bf16)p1;
    }
    // relayout P (C-layout -> A-layout) via per-wave LDS tile, then PV
    bf16x16 pf = load_a_frag(myp, 32, lane);
#pragma unroll
    for (int t = 0; t < 4; ++t) acc[t] = WMMA_BF16(pf, vf[t], acc[t]);
  }
#pragma unroll
  for (int r = 0; r < 8; ++r) {
    float inv = 1.0f / lrow[r];
    int row = q0 + r + 8 * hi;
#pragma unroll
    for (int t = 0; t < 4; ++t) {
      ctx[(size_t)row * CEMB + h * DHEAD + t * 16 + lo] = (bf16)(acc[t][r] * inv);
    }
  }
}

// ---------------- O-projection + residual (onto post-LN1 h) ----------------
__global__ __launch_bounds__(256) void oproj_kernel(const bf16* __restrict__ CTX,
                                                    const bf16* __restrict__ WoT,
                                                    const float* __restrict__ bo,
                                                    const float* __restrict__ h1f,
                                                    float* __restrict__ x2) {
  int lane = threadIdx.x & 31;
  int w = blockIdx.x * 8 + (threadIdx.x >> 5);
  const int ntn = CEMB / 64;  // 12
  int m0 = (w / ntn) * 32;
  int n0 = (w % ntn) * 64;
  v8f acc[8];
  zero_acc8(acc);
  gemm_tile_32x64<CEMB>(CTX, WoT, m0, n0, lane, acc);
  int hi = lane >> 4, lo = lane & 15;
#pragma unroll
  for (int ms = 0; ms < 2; ++ms) {
#pragma unroll
    for (int t = 0; t < 4; ++t) {
#pragma unroll
      for (int r = 0; r < 8; ++r) {
        int row = m0 + ms * 16 + r + 8 * hi;
        int n = n0 + t * 16 + lo;
        x2[(size_t)row * CEMB + n] = acc[ms * 4 + t][r] + bo[n] + h1f[(size_t)row * CEMB + n];
      }
    }
  }
}

// ---------------- FFN1: h2 @ W1 + b1, exact GELU -> bf16 ----------------
__global__ __launch_bounds__(256) void ffn1_kernel(const bf16* __restrict__ H2,
                                                   const bf16* __restrict__ W1T,
                                                   const float* __restrict__ b1,
                                                   bf16* __restrict__ M1) {
  int lane = threadIdx.x & 31;
  int w = blockIdx.x * 8 + (threadIdx.x >> 5);
  const int ntn = FFND / 64;  // 48
  int m0 = (w / ntn) * 32;
  int n0 = (w % ntn) * 64;
  v8f acc[8];
  zero_acc8(acc);
  gemm_tile_32x64<CEMB>(H2, W1T, m0, n0, lane, acc);
  int hi = lane >> 4, lo = lane & 15;
#pragma unroll
  for (int ms = 0; ms < 2; ++ms) {
#pragma unroll
    for (int t = 0; t < 4; ++t) {
#pragma unroll
      for (int r = 0; r < 8; ++r) {
        int row = m0 + ms * 16 + r + 8 * hi;
        int n = n0 + t * 16 + lo;
        float v = acc[ms * 4 + t][r] + b1[n];
        float g = 0.5f * v * (1.0f + erff(v * 0.70710678118654752440f));
        M1[(size_t)row * FFND + n] = (bf16)g;
      }
    }
  }
}

// ---------------- FFN2: m1 @ W2 + b2 + h2 -> out fp32 ----------------
__global__ __launch_bounds__(256) void ffn2_kernel(const bf16* __restrict__ M1,
                                                   const bf16* __restrict__ W2T,
                                                   const float* __restrict__ b2,
                                                   const float* __restrict__ h2f,
                                                   float* __restrict__ out) {
  int lane = threadIdx.x & 31;
  int w = blockIdx.x * 8 + (threadIdx.x >> 5);
  const int ntn = CEMB / 64;  // 12
  int m0 = (w / ntn) * 32;
  int n0 = (w % ntn) * 64;
  v8f acc[8];
  zero_acc8(acc);
  gemm_tile_32x64<FFND>(M1, W2T, m0, n0, lane, acc);
  int hi = lane >> 4, lo = lane & 15;
#pragma unroll
  for (int ms = 0; ms < 2; ++ms) {
#pragma unroll
    for (int t = 0; t < 4; ++t) {
#pragma unroll
      for (int r = 0; r < 8; ++r) {
        int row = m0 + ms * 16 + r + 8 * hi;
        int n = n0 + t * 16 + lo;
        out[(size_t)row * CEMB + n] = acc[ms * 4 + t][r] + b2[n] + h2f[(size_t)row * CEMB + n];
      }
    }
  }
}

// ---------------- host launcher ----------------
extern "C" void kernel_launch(void* const* d_in, const int* in_sizes, int n_in,
                              void* d_out, int out_size, void* d_ws, size_t ws_size,
                              hipStream_t stream) {
  const float* x    = (const float*)d_in[0];
  const float* Wq   = (const float*)d_in[1];
  const float* bq   = (const float*)d_in[2];
  const float* Wk   = (const float*)d_in[3];
  const float* bk   = (const float*)d_in[4];
  const float* Wv   = (const float*)d_in[5];
  const float* bv   = (const float*)d_in[6];
  const float* Wo   = (const float*)d_in[7];
  const float* bo   = (const float*)d_in[8];
  const float* ln1g = (const float*)d_in[9];
  const float* ln1b = (const float*)d_in[10];
  const float* ln2g = (const float*)d_in[11];
  const float* ln2b = (const float*)d_in[12];
  const float* W1   = (const float*)d_in[13];
  const float* b1   = (const float*)d_in[14];
  const float* W2   = (const float*)d_in[15];
  const float* b2   = (const float*)d_in[16];
  float* out = (float*)d_out;

  char* ws = (char*)d_ws;
  size_t off = 0;
  auto alloc = [&](size_t bytes) -> char* {
    char* p = ws + off;
    off += (bytes + 255) & ~(size_t)255;
    return p;
  };

  bf16* WqkvT = (bf16*)alloc((size_t)3 * CEMB * CEMB * 2);
  bf16* WoT   = (bf16*)alloc((size_t)CEMB * CEMB * 2);
  bf16* W1T   = (bf16*)alloc((size_t)FFND * CEMB * 2);
  bf16* W2T   = (bf16*)alloc((size_t)CEMB * FFND * 2);
  float* h1f  = (float*)alloc((size_t)TSEQ * CEMB * 4);
  bf16* h1b   = (bf16*)alloc((size_t)TSEQ * CEMB * 2);
  bf16* Qb    = (bf16*)alloc((size_t)NHEADS * TSEQ * DHEAD * 2);
  bf16* Kb    = (bf16*)alloc((size_t)NHEADS * TSEQ * DHEAD * 2);
  bf16* VTb   = (bf16*)alloc((size_t)NHEADS * DHEAD * TSEQ * 2);
  bf16* ctxb  = (bf16*)alloc((size_t)TSEQ * CEMB * 2);
  float* x2f  = (float*)alloc((size_t)TSEQ * CEMB * 4);
  float* h2f  = (float*)alloc((size_t)TSEQ * CEMB * 4);
  bf16* h2b   = (bf16*)alloc((size_t)TSEQ * CEMB * 2);
  bf16* m1b   = (bf16*)alloc((size_t)TSEQ * FFND * 2);
  (void)ws_size; (void)in_sizes; (void)n_in; (void)out_size;

  // weight conversion (transpose + bf16)
  int nsq = CEMB * CEMB;
  convert_wT<<<(nsq + 255) / 256, 256, 0, stream>>>(Wq, WqkvT, CEMB, CEMB);
  convert_wT<<<(nsq + 255) / 256, 256, 0, stream>>>(Wk, WqkvT + (size_t)CEMB * CEMB, CEMB, CEMB);
  convert_wT<<<(nsq + 255) / 256, 256, 0, stream>>>(Wv, WqkvT + (size_t)2 * CEMB * CEMB, CEMB, CEMB);
  convert_wT<<<(nsq + 255) / 256, 256, 0, stream>>>(Wo, WoT, CEMB, CEMB);
  int nw1 = CEMB * FFND;
  convert_wT<<<(nw1 + 255) / 256, 256, 0, stream>>>(W1, W1T, CEMB, FFND);
  convert_wT<<<(nw1 + 255) / 256, 256, 0, stream>>>(W2, W2T, FFND, CEMB);

  // LN1
  ln_kernel<<<TSEQ / 8, 256, 0, stream>>>(x, ln1g, ln1b, h1f, h1b);

  // QKV projection: 128 m-tiles x 36 n-tiles = 4608 waves
  qkv_kernel<<<(128 * 36) / 8, 256, 0, stream>>>(h1b, WqkvT, bq, bk, bv, Qb, Kb, VTb);

  // flash attention: 12 heads x 256 query tiles = 3072 waves
  attn_kernel<<<(NHEADS * 256) / 8, 256, 0, stream>>>(Qb, Kb, VTb, ctxb);

  // O projection + residual (onto post-LN1 h)
  oproj_kernel<<<(128 * 12) / 8, 256, 0, stream>>>(ctxb, WoT, bo, h1f, x2f);

  // LN2
  ln_kernel<<<TSEQ / 8, 256, 0, stream>>>(x2f, ln2g, ln2b, h2f, h2b);

  // FFN
  ffn1_kernel<<<(128 * 48) / 8, 256, 0, stream>>>(h2b, W1T, b1, m1b);
  ffn2_kernel<<<(128 * 12) / 8, 256, 0, stream>>>(m1b, W2T, b2, h2f, out);
}
